// AttentionNet_62440234549894
// MI455X (gfx1250) — matrix operand order repaired and checked by reading, not compiled, not run on hardware
//
#include <hip/hip_runtime.h>
#include <hip/hip_bf16.h>
#include <math.h>

// ---------------------------------------------------------------------------
// Types for WMMA bf16 fragments (wave32, v_wmma_f32_16x16x32_bf16)
// ---------------------------------------------------------------------------
typedef __attribute__((ext_vector_type(16))) __bf16 v16bf;
typedef __attribute__((ext_vector_type(8)))  __bf16 v8bf;
typedef __attribute__((ext_vector_type(8)))  float  v8f;

union BF16x16 { v16bf v; v8bf h[2]; unsigned short s[16]; };

// ---------------------------------------------------------------------------
// Problem constants
// ---------------------------------------------------------------------------
#define TT 32
#define BB 16
#define H1 52
#define W1 40
#define HW1 2080
#define H2 27
#define W2 20
#define HW2 540
#define N1 (BB*HW1)   // 33280
#define K1 192
#define M1 32
#define N2 (BB*HW2)   // 8640
#define K2 512
#define M2 64
#define K3 1728       // 192 ch * 3 * 3
#define M3 512        // 4 * 128 gates
#define HID 256
#define NQ 4
#define NKEY 72
#define NVAL 184
#define NACT 18
#define ANSK 1043
#define ANSKP 1056    // padded to multiple of 32
#define PI_F 3.14159265358979323846f

// ---------------------------------------------------------------------------
// Scalar helpers
// ---------------------------------------------------------------------------
static __device__ __forceinline__ unsigned short f2bf(float f) {
  union { float f; unsigned int u; } v; v.f = f;
  unsigned int r = (v.u + 0x7FFFu + ((v.u >> 16) & 1u)) >> 16;
  return (unsigned short)r;
}
static __device__ __forceinline__ float bf2f(unsigned short h) {
  union { unsigned int u; float f; } v; v.u = ((unsigned int)h) << 16;
  return v.f;
}
static __device__ __forceinline__ float sigm(float x) {
  return 1.0f / (1.0f + __expf(-x));
}

// ---------------------------------------------------------------------------
// Weight conversion kernels (f32 -> bf16; plain and transpose+pad)
// ---------------------------------------------------------------------------
__global__ void k_cvt(const float* __restrict__ in, unsigned short* __restrict__ out, int n) {
  int i = blockIdx.x * blockDim.x + threadIdx.x;
  if (i < n) out[i] = f2bf(in[i]);
}

// in is [K][M] row-major; out is [Mp][Kp] row-major bf16, zero padded
__global__ void k_cvt_t(const float* __restrict__ in, unsigned short* __restrict__ out,
                        int M, int K, int Mp, int Kp) {
  int i = blockIdx.x * blockDim.x + threadIdx.x;
  if (i >= Mp * Kp) return;
  int k = i % Kp, m = i / Kp;
  float v = (m < M && k < K) ? in[(size_t)k * M + m] : 0.0f;
  out[i] = f2bf(v);
}

// ---------------------------------------------------------------------------
// Spatial cosine basis S[540][64]
// ---------------------------------------------------------------------------
__global__ void k_sbasis(float* __restrict__ S) {
  int i = blockIdx.x * blockDim.x + threadIdx.x;
  if (i >= HW2 * 64) return;
  int uv = i & 63, p = i >> 6;
  int u = (uv >> 3) + 1, v = (uv & 7) + 1;
  int y = p / W2, x = p % W2;
  float ph = (float)(y + 1) * (PI_F / (float)H2);
  float pw = (float)(x + 1) * (PI_F / (float)W2);
  S[i] = __cosf(ph * (float)u) * __cosf(pw * (float)v);
}

// ---------------------------------------------------------------------------
// im2col kernels (bf16 patch matrices)
// ---------------------------------------------------------------------------
__global__ void k_im2col_conv1(const float* __restrict__ frame_t, // (16,3,210,160)
                               unsigned short* __restrict__ P1) { // [N1][K1]
  int idx = blockIdx.x * blockDim.x + threadIdx.x;
  if (idx >= N1 * K1) return;
  int k = idx % K1, n = idx / K1;
  int c = k >> 6, rem = k & 63, ky = rem >> 3, kx = rem & 7;
  int ox = n % W1; int t1 = n / W1; int oy = t1 % H1; int img = t1 / H1;
  int iy = oy * 4 - 1 + ky, ix = ox * 4 - 2 + kx;
  float v = 0.0f;
  if (iy >= 0 && iy < 210 && ix >= 0 && ix < 160)
    v = frame_t[((size_t)(img * 3 + c) * 210 + iy) * 160 + ix];
  P1[idx] = f2bf(v);
}

__global__ void k_im2col_conv2(const unsigned short* __restrict__ c1, // (16,32,52,40) bf16
                               unsigned short* __restrict__ P2) {     // [N2][K2]
  int idx = blockIdx.x * blockDim.x + threadIdx.x;
  if (idx >= N2 * K2) return;
  int k = idx % K2, n = idx / K2;
  int c = k >> 4, ky = (k >> 2) & 3, kx = k & 3;
  int ox = n % W2; int t1 = n / W2; int oy = t1 % H2; int img = t1 / H2;
  int iy = oy * 2 - 2 + ky, ix = ox * 2 - 1 + kx;
  unsigned short v = 0;
  if (iy >= 0 && iy < H1 && ix >= 0 && ix < W1)
    v = c1[((size_t)(img * M1 + c) * H1 + iy) * W1 + ix];
  P2[idx] = v;
}

__global__ void k_im2col_lstm(const unsigned short* __restrict__ xb, // (16,64,540) bf16
                              const float* __restrict__ hst,          // (16,128,540)
                              const unsigned char* __restrict__ done_t,
                              unsigned short* __restrict__ P3) {      // [N2][K3]
  int idx = blockIdx.x * blockDim.x + threadIdx.x;
  if (idx >= N2 * K3) return;
  int k = idx % K3, n = idx / K3;
  int c = k / 9; int r9 = k % 9; int ky = r9 / 3, kx = r9 % 3;
  int ox = n % W2; int t1 = n / W2; int oy = t1 % H2; int b = t1 / H2;
  int iy = oy - 1 + ky, ix = ox - 1 + kx;
  unsigned short v = 0;
  if (iy >= 0 && iy < H2 && ix >= 0 && ix < W2) {
    int p = iy * W2 + ix;
    if (c < 64) {
      v = xb[((size_t)b * 64 + c) * HW2 + p];
    } else {
      float nd = done_t[b] ? 0.0f : 1.0f;
      v = f2bf(hst[((size_t)b * 128 + (c - 64)) * HW2 + p] * nd);
    }
  }
  P3[idx] = v;
}

// ---------------------------------------------------------------------------
// WMMA GEMM with NCHW epilogue, register-blocked on M:
//   acc[n,m] = sum_k P[n][k]*W[m][k] + bias[m];  n = img*HW + p;
//   out[(img*M + m)*HW + p]
// Block: 256 threads = 8 waves (4 n-tiles x 2 m-groups).
// Each wave: one A fragment reused across MB B fragments / accumulators.
// Block covers 64 n x (2*MB*16) m.  K%32, N%64, M%(32*MB) must be exact.
// OUTF selects f32 output (true) vs bf16 output (false) at compile time.
// ---------------------------------------------------------------------------
template <int MB, bool OUTF>
__global__ void __launch_bounds__(256) k_gemm_nchw(
    const unsigned short* __restrict__ P, const unsigned short* __restrict__ W,
    const float* __restrict__ bias,
    unsigned short* __restrict__ outB, float* __restrict__ outF,
    int N, int M, int K, int HW) {
  int lane = threadIdx.x & 31;
  int wave = threadIdx.x >> 5;
  int nt  = blockIdx.x * 64 + (wave & 3) * 16;
  int mt0 = blockIdx.y * (2 * MB * 16) + (wave >> 2) * (MB * 16);
  int r  = lane & 15;
  int hi = lane >> 4;
  const unsigned short* prow = P + (size_t)(nt + r) * K + hi * 8;    // A chunks: k0, k0+16
  const unsigned short* wrow = W + (size_t)(mt0 + r) * K + hi * 8;   // B chunks: k0, k0+16
  v8f acc[MB];
#pragma unroll
  for (int i = 0; i < MB; ++i) acc[i] = v8f{};
  for (int k0 = 0; k0 < K; k0 += 32) {
    // prefetch the streaming A operand one K-tile ahead (global_prefetch_b8)
    if (k0 + 32 < K) __builtin_prefetch(prow + k0 + 32, 0, 3);
    BF16x16 a;
    a.h[0] = *(const v8bf*)(prow + k0);
    a.h[1] = *(const v8bf*)(prow + k0 + 16);
#pragma unroll
    for (int i = 0; i < MB; ++i) {
      BF16x16 b;
      b.h[0] = *(const v8bf*)(wrow + (size_t)i * 16 * K + k0);
      b.h[1] = *(const v8bf*)(wrow + (size_t)i * 16 * K + k0 + 16);
      acc[i] = __builtin_amdgcn_wmma_f32_16x16x32_bf16(false, a.v, false, b.v,
                                                       (short)0, acc[i], false, false);
    }
  }
#pragma unroll
  for (int i = 0; i < MB; ++i) {
    int m = mt0 + i * 16 + r;
    float bv = bias[m];
#pragma unroll
    for (int v = 0; v < 8; ++v) {
      int n = nt + v + hi * 8;
      int img = n / HW, p = n - img * HW;
      size_t o = ((size_t)img * M + m) * HW + p;
      float val = acc[i][v] + bv;
      if (OUTF) outF[o] = val;
      else      outB[o] = f2bf(val);
    }
  }
}

// ---------------------------------------------------------------------------
// ConvLSTM pointwise update; writes h into O[t] (T,B,540,128) bf16
// ---------------------------------------------------------------------------
__global__ void k_lstm_point(const float* __restrict__ gates, // (16,512,540)
                             float* __restrict__ hst, float* __restrict__ cst, // (16,128,540)
                             const unsigned char* __restrict__ done_t,
                             unsigned short* __restrict__ O, int t) {
  int idx = blockIdx.x * blockDim.x + threadIdx.x;
  if (idx >= BB * 128 * HW2) return;
  int p = idx % HW2; int ch = (idx / HW2) % 128; int b = idx / (HW2 * 128);
  size_t gb = (size_t)b * M3 * HW2 + p;
  float ai = gates[gb + (size_t)(ch)       * HW2];
  float af = gates[gb + (size_t)(128 + ch) * HW2];
  float ao = gates[gb + (size_t)(256 + ch) * HW2];
  float ag = gates[gb + (size_t)(384 + ch) * HW2];
  float nd = done_t[b] ? 0.0f : 1.0f;
  size_t si = ((size_t)b * 128 + ch) * HW2 + p;
  float cp = cst[si] * nd;
  float cn = sigm(af) * cp + sigm(ai) * tanhf(ag);
  float hn = sigm(ao) * tanhf(cn);
  cst[si] = cn;
  hst[si] = hn;
  O[(((size_t)t * BB + b) * HW2 + p) * 128 + ch] = f2bf(hn);
}

// ---------------------------------------------------------------------------
// Core recurrent kernel: one workgroup (8 waves), loops t internally.
// Batch(16) == WMMA M-tile.  A operand comes pre-converted as bf16 [16][ldA];
// each stage dual-writes f32 (for VALU consumers) + bf16 (next GEMM's A).
// RELU/ACCUM are compile-time so the epilogue is branch-free; bias mandatory.
// ---------------------------------------------------------------------------
template <bool RELU, bool ACCUM>
static __device__ void wave_gemm16(const unsigned short* actb, int ldA,
                                   const unsigned short* Wt, const float* bias,
                                   float* out, unsigned short* outb,
                                   int ldO, int N, int K) {
  int lane = threadIdx.x & 31;
  int wave = threadIdx.x >> 5;
  int r = lane & 15, hi = lane >> 4;
  const unsigned short* arow = actb + (size_t)r * ldA + hi * 8;
  for (int ntile = wave * 16; ntile < N; ntile += 8 * 16) {
    const unsigned short* wrow = Wt + (size_t)(ntile + r) * K + hi * 8;
    v8f acc = {};
    for (int k0 = 0; k0 < K; k0 += 32) {
      BF16x16 a, b;
      a.h[0] = *(const v8bf*)(arow + k0);
      a.h[1] = *(const v8bf*)(arow + k0 + 16);
      b.h[0] = *(const v8bf*)(wrow + k0);
      b.h[1] = *(const v8bf*)(wrow + k0 + 16);
      acc = __builtin_amdgcn_wmma_f32_16x16x32_bf16(false, a.v, false, b.v,
                                                    (short)0, acc, false, false);
    }
    int col = ntile + r;
    float bv = bias[col];
#pragma unroll
    for (int v = 0; v < 8; ++v) {
      int row = v + hi * 8;           // batch
      float val = acc[v] + bv;
      if (ACCUM) val += out[(size_t)row * ldO + col];
      if (RELU)  val = fmaxf(val, 0.0f);
      out[(size_t)row * ldO + col]  = val;
      outb[(size_t)row * ldO + col] = f2bf(val);
    }
  }
}

#define CORE_SYNC() do { __threadfence(); __syncthreads(); } while (0)

__global__ void __launch_bounds__(256) k_core(
    const unsigned short* __restrict__ O, const float* __restrict__ S,
    const unsigned char* __restrict__ done, const int* __restrict__ last_action,
    const float* __restrict__ reward,
    const float* __restrict__ core_h0, const float* __restrict__ core_c0,
    const unsigned short* __restrict__ qw1t, const float* __restrict__ qb1,
    const unsigned short* __restrict__ qw2t, const float* __restrict__ qb2,
    const unsigned short* __restrict__ qw3t, const float* __restrict__ qb3,
    const unsigned short* __restrict__ aw1t, const float* __restrict__ ab1,
    const unsigned short* __restrict__ aw2t, const float* __restrict__ ab2,
    const unsigned short* __restrict__ wihc, const float* __restrict__ b_ih,
    const unsigned short* __restrict__ whhc, const float* __restrict__ b_hh,
    float* __restrict__ attnA,          // (16,540,4) f32
    float* __restrict__ stageA,         // 16x288 f32
    unsigned short* __restrict__ stageAb,
    float* __restrict__ qbuf,           // 16x288 f32
    unsigned short* __restrict__ qbufb,
    unsigned short* __restrict__ answb, // 16x1056 bf16
    float* __restrict__ a1buf,          // 16x512 f32
    unsigned short* __restrict__ a1b,
    float* __restrict__ cibuf,          // 16x256 f32
    unsigned short* __restrict__ cib,
    float* __restrict__ gate2,          // 16x1024 f32
    unsigned short* __restrict__ gate2b,
    unsigned short* __restrict__ hb,    // 16x256 bf16 (masked h)
    float* __restrict__ outs,           // (32,16,256)
    float* __restrict__ ch_out, float* __restrict__ cc_out) {
  __shared__ float h_s[BB][HID];
  __shared__ float c_s[BB][HID];
  int tid = threadIdx.x;
  int lane = tid & 31, wave = tid >> 5;

  for (int i = tid; i < BB * HID; i += 256) {
    h_s[i / HID][i % HID] = core_h0[i];
    c_s[i / HID][i % HID] = core_c0[i];
  }
  __syncthreads();

  for (int t = 0; t < TT; ++t) {
    // mask state at step start; also publish h as bf16 GEMM operand
    for (int i = tid; i < BB * HID; i += 256) {
      int b = i / HID, j = i % HID;
      float nd = done[t * BB + b] ? 0.0f : 1.0f;
      float hm = h_s[b][j] * nd;
      h_s[b][j] = hm;
      c_s[b][j] *= nd;
      hb[i] = f2bf(hm);
    }
    CORE_SYNC();
    // q-MLP: h -> 128 -> 288 -> 288
    wave_gemm16<true, false>(hb, HID, qw1t, qb1, stageA, stageAb, 128, 128, HID);
    CORE_SYNC();
    wave_gemm16<true, false>(stageAb, 128, qw2t, qb2, qbuf, qbufb, 288, 288, 128);
    CORE_SYNC();
    wave_gemm16<false, false>(qbufb, 288, qw3t, qb3, stageA, stageAb, 288, 288, 288);
    CORE_SYNC();
    // attention logits: A[b,p,qi] = sum_k K_t[b,p,k]*q[b,qi,k]
    for (int i = tid; i < BB * HW2 * NQ; i += 256) {
      int qi = i & 3; int p = (i >> 2) % HW2; int b = i / (HW2 * NQ);
      const unsigned short* Op = O + (((size_t)t * BB + b) * HW2 + p) * 128;
      const float* qv = stageA + b * 288 + qi * NKEY;
      float s = 0.0f;
      for (int k = 0; k < 8; ++k)  s += bf2f(Op[k]) * qv[k];
      for (int k = 0; k < 64; ++k) s += S[p * 64 + k] * qv[8 + k];
      attnA[((size_t)b * HW2 + p) * NQ + qi] = s;
    }
    CORE_SYNC();
    // softmax over p for each (b,qi); one wave per group
    for (int g = wave; g < BB * NQ; g += 8) {
      int b = g >> 2, qi = g & 3;
      float mx = -1e30f;
      for (int p = lane; p < HW2; p += 32)
        mx = fmaxf(mx, attnA[((size_t)b * HW2 + p) * NQ + qi]);
      for (int o = 16; o > 0; o >>= 1) mx = fmaxf(mx, __shfl_xor(mx, o, 32));
      float sum = 0.0f;
      for (int p = lane; p < HW2; p += 32)
        sum += __expf(attnA[((size_t)b * HW2 + p) * NQ + qi] - mx);
      for (int o = 16; o > 0; o >>= 1) sum += __shfl_xor(sum, o, 32);
      float inv = 1.0f / sum;
      for (int p = lane; p < HW2; p += 32) {
        size_t ix = ((size_t)b * HW2 + p) * NQ + qi;
        attnA[ix] = __expf(attnA[ix] - mx) * inv;
      }
    }
    CORE_SYNC();
    // ans[b,qi,v] = sum_p A[b,p,qi]*V[b,p,v]  -> answer[:,0:736] (bf16)
    for (int i = tid; i < BB * NQ * NVAL; i += 256) {
      int v = i % NVAL; int qi = (i / NVAL) & 3; int b = i / (NVAL * NQ);
      float s = 0.0f;
      for (int p = 0; p < HW2; ++p) {
        float a = attnA[((size_t)b * HW2 + p) * NQ + qi];
        float val = (v < 120)
            ? bf2f(O[(((size_t)t * BB + b) * HW2 + p) * 128 + 8 + v])
            : S[p * 64 + (v - 120)];
        s += a * val;
      }
      answb[(size_t)b * ANSKP + qi * NVAL + v] = f2bf(s);
    }
    // fill answer tail: q(288) | r(1) | onehot(18) | pad->1056
    for (int i = tid; i < BB * (ANSKP - NQ * NVAL); i += 256) {
      int b = i / (ANSKP - NQ * NVAL); int j = i % (ANSKP - NQ * NVAL);
      int pos = NQ * NVAL + j;
      float val;
      if (j < 288) val = stageA[b * 288 + j];
      else if (pos == 1024) val = fminf(fmaxf(reward[t * BB + b], -1.0f), 1.0f);
      else if (pos >= 1025 && pos < 1025 + NACT)
        val = (last_action[t * BB + b] == pos - 1025) ? 1.0f : 0.0f;
      else val = 0.0f;
      answb[(size_t)b * ANSKP + pos] = f2bf(val);
    }
    CORE_SYNC();
    // answer-MLP: 1056 -> 512 (relu) -> 256
    wave_gemm16<true, false>(answb, ANSKP, aw1t, ab1, a1buf, a1b, 512, 512, ANSKP);
    CORE_SYNC();
    wave_gemm16<false, false>(a1b, 512, aw2t, ab2, cibuf, cib, HID, HID, 512);
    CORE_SYNC();
    // gates = ci@Wih^T + b_ih + h@Whh^T + b_hh
    wave_gemm16<false, false>(cib, HID, wihc, b_ih, gate2, gate2b, 1024, 1024, HID);
    CORE_SYNC();
    wave_gemm16<false, true>(hb, HID, whhc, b_hh, gate2, gate2b, 1024, 1024, HID);
    CORE_SYNC();
    // LSTM pointwise
    for (int i = tid; i < BB * HID; i += 256) {
      int b = i / HID, j = i % HID;
      float gi = gate2[(size_t)b * 1024 + j];
      float gf = gate2[(size_t)b * 1024 + 256 + j];
      float gg = gate2[(size_t)b * 1024 + 512 + j];
      float go = gate2[(size_t)b * 1024 + 768 + j];
      float cn = sigm(gf) * c_s[b][j] + sigm(gi) * tanhf(gg);
      float hn = sigm(go) * tanhf(cn);
      c_s[b][j] = cn;
      h_s[b][j] = hn;
      outs[((size_t)t * BB + b) * HID + j] = hn;
    }
    CORE_SYNC();
  }
  for (int i = tid; i < BB * HID; i += 256) {
    ch_out[i] = h_s[i / HID][i % HID];
    cc_out[i] = c_s[i / HID][i % HID];
  }
}

// ---------------------------------------------------------------------------
// Heads: logits, values, argmax (512 rows x 256)
// ---------------------------------------------------------------------------
__global__ void k_heads(const float* __restrict__ outs,
                        const float* __restrict__ pw, const float* __restrict__ pb,
                        const float* __restrict__ vw, const float* __restrict__ vb,
                        float* __restrict__ logits, float* __restrict__ values,
                        float* __restrict__ action) {
  int row = blockIdx.x * blockDim.x + threadIdx.x;
  if (row >= TT * BB) return;
  const float* x = outs + (size_t)row * HID;
  float best = -1e30f; int bi = 0;
  for (int a = 0; a < NACT; ++a) {
    float s = pb[a];
    for (int k = 0; k < HID; ++k) s += x[k] * pw[k * NACT + a];
    logits[(size_t)row * NACT + a] = s;
    if (s > best) { best = s; bi = a; }
  }
  float v = vb[0];
  for (int k = 0; k < HID; ++k) v += x[k] * vw[k];
  values[row] = v;
  action[row] = (float)bi;
}

// ---------------------------------------------------------------------------
// Host launch
// ---------------------------------------------------------------------------
extern "C" void kernel_launch(void* const* d_in, const int* in_sizes, int n_in,
                              void* d_out, int out_size, void* d_ws, size_t ws_size,
                              hipStream_t stream) {
  (void)in_sizes; (void)n_in; (void)out_size; (void)ws_size;
  const float* frame        = (const float*)d_in[0];
  const unsigned char* done = (const unsigned char*)d_in[1];
  const int* last_action    = (const int*)d_in[2];
  const float* reward       = (const float*)d_in[3];
  const float* core_h0      = (const float*)d_in[4];
  const float* core_c0      = (const float*)d_in[5];
  const float* conv_h0      = (const float*)d_in[6];
  const float* conv_c0      = (const float*)d_in[7];
  const float* cnn_w1 = (const float*)d_in[8];   const float* cnn_b1 = (const float*)d_in[9];
  const float* cnn_w2 = (const float*)d_in[10];  const float* cnn_b2 = (const float*)d_in[11];
  const float* lstm_w = (const float*)d_in[12];  const float* lstm_b = (const float*)d_in[13];
  const float* qw1 = (const float*)d_in[14];     const float* qb1 = (const float*)d_in[15];
  const float* qw2 = (const float*)d_in[16];     const float* qb2 = (const float*)d_in[17];
  const float* qw3 = (const float*)d_in[18];     const float* qb3 = (const float*)d_in[19];
  const float* aw1 = (const float*)d_in[20];     const float* ab1 = (const float*)d_in[21];
  const float* aw2 = (const float*)d_in[22];     const float* ab2 = (const float*)d_in[23];
  const float* w_ih = (const float*)d_in[24];    const float* w_hh = (const float*)d_in[25];
  const float* b_ih = (const float*)d_in[26];    const float* b_hh = (const float*)d_in[27];
  const float* pw = (const float*)d_in[28];      const float* pb = (const float*)d_in[29];
  const float* vw = (const float*)d_in[30];      const float* vb = (const float*)d_in[31];
  float* dout = (float*)d_out;

  // --- carve workspace ---
  char* base = (char*)d_ws; size_t off = 0;
  auto alloc = [&](size_t bytes) -> void* {
    void* p = base + off; off = (off + bytes + 255) & ~(size_t)255; return p;
  };
  unsigned short* w1c  = (unsigned short*)alloc((size_t)M1 * K1 * 2);
  unsigned short* w2c  = (unsigned short*)alloc((size_t)M2 * K2 * 2);
  unsigned short* wl   = (unsigned short*)alloc((size_t)M3 * K3 * 2);
  unsigned short* qw1t = (unsigned short*)alloc((size_t)128 * 256 * 2);
  unsigned short* qw2t = (unsigned short*)alloc((size_t)288 * 128 * 2);
  unsigned short* qw3t = (unsigned short*)alloc((size_t)288 * 288 * 2);
  unsigned short* aw1t = (unsigned short*)alloc((size_t)512 * ANSKP * 2);
  unsigned short* aw2t = (unsigned short*)alloc((size_t)256 * 512 * 2);
  unsigned short* wihc = (unsigned short*)alloc((size_t)1024 * 256 * 2);
  unsigned short* whhc = (unsigned short*)alloc((size_t)1024 * 256 * 2);
  float* Sbuf          = (float*)alloc((size_t)HW2 * 64 * 4);
  unsigned short* P1   = (unsigned short*)alloc((size_t)N1 * K1 * 2);
  unsigned short* c1   = (unsigned short*)alloc((size_t)BB * M1 * HW1 * 2);
  unsigned short* P2   = (unsigned short*)alloc((size_t)N2 * K2 * 2);
  unsigned short* xb   = (unsigned short*)alloc((size_t)BB * M2 * HW2 * 2);
  unsigned short* P3   = (unsigned short*)alloc((size_t)N2 * K3 * 2);
  float* gates         = (float*)alloc((size_t)BB * M3 * HW2 * 4);
  float* hst           = (float*)alloc((size_t)BB * 128 * HW2 * 4);
  float* cst           = (float*)alloc((size_t)BB * 128 * HW2 * 4);
  unsigned short* Obuf = (unsigned short*)alloc((size_t)TT * BB * HW2 * 128 * 2);
  float* attnA         = (float*)alloc((size_t)BB * HW2 * NQ * 4);
  float* stageA        = (float*)alloc((size_t)BB * 288 * 4);
  unsigned short* stageAb = (unsigned short*)alloc((size_t)BB * 288 * 2);
  float* qbuf          = (float*)alloc((size_t)BB * 288 * 4);
  unsigned short* qbufb   = (unsigned short*)alloc((size_t)BB * 288 * 2);
  unsigned short* answb   = (unsigned short*)alloc((size_t)BB * ANSKP * 2);
  float* a1buf         = (float*)alloc((size_t)BB * 512 * 4);
  unsigned short* a1b     = (unsigned short*)alloc((size_t)BB * 512 * 2);
  float* cibuf         = (float*)alloc((size_t)BB * 256 * 4);
  unsigned short* cib     = (unsigned short*)alloc((size_t)BB * 256 * 2);
  float* gate2         = (float*)alloc((size_t)BB * 1024 * 4);
  unsigned short* gate2b  = (unsigned short*)alloc((size_t)BB * 1024 * 2);
  unsigned short* hb      = (unsigned short*)alloc((size_t)BB * HID * 2);
  float* outs          = (float*)alloc((size_t)TT * BB * HID * 4);

  // --- one-time setup ---
  auto cvt = [&](const float* in, unsigned short* out, int n) {
    k_cvt<<<(n + 255) / 256, 256, 0, stream>>>(in, out, n);
  };
  cvt(cnn_w1, w1c, M1 * K1);
  cvt(cnn_w2, w2c, M2 * K2);
  cvt(lstm_w, wl, M3 * K3);
  cvt(w_ih, wihc, 1024 * 256);
  cvt(w_hh, whhc, 1024 * 256);
  k_cvt_t<<<(128 * 256 + 255) / 256, 256, 0, stream>>>(qw1, qw1t, 128, 256, 128, 256);
  k_cvt_t<<<(288 * 128 + 255) / 256, 256, 0, stream>>>(qw2, qw2t, 288, 128, 288, 128);
  k_cvt_t<<<(288 * 288 + 255) / 256, 256, 0, stream>>>(qw3, qw3t, 288, 288, 288, 288);
  k_cvt_t<<<(512 * ANSKP + 255) / 256, 256, 0, stream>>>(aw1, aw1t, 512, ANSK, 512, ANSKP);
  k_cvt_t<<<(256 * 512 + 255) / 256, 256, 0, stream>>>(aw2, aw2t, 256, 512, 256, 512);
  k_sbasis<<<(HW2 * 64 + 255) / 256, 256, 0, stream>>>(Sbuf);
  hipMemcpyAsync(hst, conv_h0, (size_t)BB * 128 * HW2 * 4, hipMemcpyDeviceToDevice, stream);
  hipMemcpyAsync(cst, conv_c0, (size_t)BB * 128 * HW2 * 4, hipMemcpyDeviceToDevice, stream);

  // --- per-timestep vision pipeline (CNN + ConvLSTM) ---
  for (int t = 0; t < TT; ++t) {
    const float* frame_t = frame + (size_t)t * BB * 3 * 210 * 160;
    const unsigned char* done_t = done + (size_t)t * BB;
    k_im2col_conv1<<<(N1 * K1) / 256, 256, 0, stream>>>(frame_t, P1);
    k_gemm_nchw<1, false><<<dim3(N1 / 64, 1), 256, 0, stream>>>(
        P1, w1c, cnn_b1, c1, nullptr, N1, M1, K1, HW1);
    k_im2col_conv2<<<(N2 * K2) / 256, 256, 0, stream>>>(c1, P2);
    k_gemm_nchw<2, false><<<dim3(N2 / 64, 1), 256, 0, stream>>>(
        P2, w2c, cnn_b2, xb, nullptr, N2, M2, K2, HW2);
    k_im2col_lstm<<<(N2 * K3) / 256, 256, 0, stream>>>(xb, hst, done_t, P3);
    k_gemm_nchw<4, true><<<dim3(N2 / 64, M3 / 128), 256, 0, stream>>>(
        P3, wl, lstm_b, nullptr, gates, N2, M3, K3, HW2);
    k_lstm_point<<<(BB * 128 * HW2) / 256, 256, 0, stream>>>(
        gates, hst, cst, done_t, Obuf, t);
  }

  // --- recurrent attention core (single persistent workgroup) ---
  k_core<<<1, 256, 0, stream>>>(
      Obuf, Sbuf, done, last_action, reward, core_h0, core_c0,
      qw1t, qb1, qw2t, qb2, qw3t, qb3, aw1t, ab1, aw2t, ab2,
      wihc, b_ih, whhc, b_hh,
      attnA, stageA, stageAb, qbuf, qbufb, answb,
      a1buf, a1b, cibuf, cib, gate2, gate2b, hb, outs,
      dout + 10240, dout + 14336);

  // --- heads + final states ---
  k_heads<<<(TT * BB + 255) / 256, 256, 0, stream>>>(
      outs, pw, pb, vw, vb, dout, dout + 9216, dout + 9728);
  hipMemcpyAsync(dout + 18432, hst, (size_t)BB * 128 * HW2 * 4,
                 hipMemcpyDeviceToDevice, stream);
  hipMemcpyAsync(dout + 18432 + (size_t)BB * 128 * HW2, cst,
                 (size_t)BB * 128 * HW2 * 4, hipMemcpyDeviceToDevice, stream);
}
